// BasicLayer_49598282334858
// MI455X (gfx1250) — matrix operand order repaired and checked by reading, not compiled, not run on hardware
//
#include <hip/hip_runtime.h>
#include <math.h>

typedef __attribute__((ext_vector_type(16))) _Float16 v16h;
typedef __attribute__((ext_vector_type(8)))  float    v8f;
typedef __attribute__((ext_vector_type(4)))  unsigned u32x4;
typedef _Float16 h16;

#define C_DIM 192
#define N_HEADS 6
#define NTOK 98
#define NPAD 112
#define NWIN 1024
#define TOKS 100352
#define HID 768
#define MROWS (NWIN * NPAD)   // 114688

// ---- workspace byte offsets (MLP buffers alias attention buffers) ----
#define OFF_X16   ((size_t)0)           // [114688][192] f16 (aliases LN2)
#define OFF_Q     ((size_t)44040192)    // [1024][6][112][32] f16 (aliases FC1)
#define OFF_K     ((size_t)88080384)
#define OFF_V     ((size_t)132120576)
#define OFF_AO    ((size_t)176160768)   // [1024][112][192] f16
#define OFF_LN2   ((size_t)0)           // [100352][192] f16
#define OFF_FC1   ((size_t)44040192)    // [100352][768] f16
#define OFF_WQKV  ((size_t)220200960)   // swizzled [K/32][N][2][16] f16
#define OFF_WPROJ (OFF_WQKV + 221184)
#define OFF_WFC1  (OFF_WPROJ + 73728)
#define OFF_WFC2  (OFF_WFC1 + 294912)

// ================= vectorized WMMA fragment loaders ================================
union Frag { u32x4 q[2]; v16h h; };

__device__ __forceinline__ v16h frag_2x16B(const h16* p0, const h16* p1) {
  Frag f;
  f.q[0] = *(const u32x4*)p0;
  f.q[1] = *(const u32x4*)p1;
  return f.h;
}
// A 16x32 (MxK) fragment from row-major storage (leading dim ld, 16B-aligned rows).
// Lane row = lane&15, K split {0-15} / {16-31} by lane half (this code's contiguous order; B matches).
__device__ __forceinline__ v16h a_frag(const h16* base, int ld, int lane) {
  const h16* r = base + (lane & 15) * ld + (lane >> 4) * 16;
  return frag_2x16B(r, r + 8);
}
// B fragment when the 16 halfs for (lane, khalf) are contiguous (32B run).
__device__ __forceinline__ v16h c_frag32(const h16* p) { return frag_2x16B(p, p + 8); }

// ======= generic GEMM inner loop: one wave computes 16 x 64, software-pipelined ====
// A row-major [M][lda]; Bsw fragment-swizzled [K/32][N][2][16].
__device__ __forceinline__ void gemm_rows16(const h16* __restrict__ A, int lda,
                                            const h16* __restrict__ Bsw, int N, int K,
                                            size_t mrow0, int n0, int lane, v8f acc[4]) {
#pragma unroll
  for (int nt = 0; nt < 4; ++nt)
#pragma unroll
    for (int r = 0; r < 8; ++r) acc[nt][r] = 0.f;
  int row = lane & 15, kh = lane >> 4;
  const h16* qA = A + (mrow0 + row) * (size_t)lda + kh * 16;
  const h16* qB = Bsw + ((size_t)(n0 + row) * 2 + kh) * 16;
  const size_t strideB = (size_t)N * 32;              // halfs per k-block
  int KB = K >> 5;
  // prologue: fragments for kb = 0
  v16h af = frag_2x16B(qA, qA + 8);
  v16h b0 = c_frag32(qB);
  v16h b1 = c_frag32(qB + 512);
  v16h b2 = c_frag32(qB + 1024);
  v16h b3 = c_frag32(qB + 1536);
  for (int kb = 0; kb < KB; ++kb) {
    v16h afn = af, b0n = b0, b1n = b1, b2n = b2, b3n = b3;
    if (kb + 1 < KB) {                                // prefetch next k-block
      qA += 32;
      qB += strideB;
      afn = frag_2x16B(qA, qA + 8);
      b0n = c_frag32(qB);
      b1n = c_frag32(qB + 512);
      b2n = c_frag32(qB + 1024);
      b3n = c_frag32(qB + 1536);
    }
    acc[0] = __builtin_amdgcn_wmma_f32_16x16x32_f16(false, af, false, b0, (short)0, acc[0], false, false);
    acc[1] = __builtin_amdgcn_wmma_f32_16x16x32_f16(false, af, false, b1, (short)0, acc[1], false, false);
    acc[2] = __builtin_amdgcn_wmma_f32_16x16x32_f16(false, af, false, b2, (short)0, acc[2], false, false);
    acc[3] = __builtin_amdgcn_wmma_f32_16x16x32_f16(false, af, false, b3, (short)0, acc[3], false, false);
    af = afn; b0 = b0n; b1 = b1n; b2 = b2n; b3 = b3n;
  }
}

// ======== weight convert + swizzle:  out[((kb*N+n)*2+kh)*16+e] = W[n*K + kb*32+kh*16+e]
__global__ void convert_wT_sw(const float* __restrict__ W, h16* __restrict__ out,
                              int K, int N) {
  int i = blockIdx.x * blockDim.x + threadIdx.x;
  if (i >= K * N) return;
  int e = i & 15;
  int t = i >> 4;
  int kh = t & 1; t >>= 1;
  int n = t % N, kb = t / N;
  int k = kb * 32 + kh * 16 + e;
  out[i] = (h16)W[(size_t)n * K + k];
}

// ================= block reduction over 192 threads ================================
__device__ __forceinline__ float block_reduce_192(float v, float* red, int c) {
  red[c] = v; __syncthreads();
  if (c < 96) red[c] += red[c + 96]; __syncthreads();
  if (c < 48) red[c] += red[c + 48]; __syncthreads();
  if (c < 24) red[c] += red[c + 24]; __syncthreads();
  if (c < 12) red[c] += red[c + 12]; __syncthreads();
  if (c < 6)  red[c] += red[c + 6];  __syncthreads();
  if (c == 0) red[0] += red[1] + red[2] + red[3] + red[4] + red[5];
  __syncthreads();
  float s = red[0];
  __syncthreads();
  return s;
}

// ================= LN1 + (shift) + window partition -> f16 =========================
__global__ __launch_bounds__(192) void ln_part_kernel(const float* __restrict__ x,
    const float* __restrict__ g, const float* __restrict__ b,
    h16* __restrict__ out, int shifted) {
  int row = blockIdx.x, c = threadIdx.x;
  int win = row / NPAD, n = row % NPAD;
  h16* orow = out + (size_t)row * C_DIM;
  if (n >= NTOK) { orow[c] = (h16)0.f; return; }
  int bb = win >> 9, wd = (win >> 6) & 7, wh = (win >> 3) & 7, ww = win & 7;
  int td = n / 49, rem = n % 49, th = rem / 7, tw = rem % 7;
  int d = wd * 2 + td, h = wh * 7 + th, w = ww * 7 + tw;
  if (shifted) { d = (d + 1) & 15; h = (h + 3) % 56; w = (w + 3) % 56; }
  size_t tok = (((size_t)bb * 16 + d) * 56 + h) * 56 + w;
  float v = x[tok * C_DIM + c];
  __shared__ float red[192];
  float mu = block_reduce_192(v, red, c) * (1.f / 192.f);
  float dv = v - mu;
  float var = block_reduce_192(dv * dv, red, c) * (1.f / 192.f);
  orow[c] = (h16)(dv * rsqrtf(var + 1e-5f) * g[c] + b[c]);
}

// ================= LN2 (plain token layout) -> f16 =================================
__global__ __launch_bounds__(192) void ln2_kernel(const float* __restrict__ x,
    const float* __restrict__ g, const float* __restrict__ b, h16* __restrict__ out) {
  size_t row = blockIdx.x; int c = threadIdx.x;
  float v = x[row * C_DIM + c];
  __shared__ float red[192];
  float mu = block_reduce_192(v, red, c) * (1.f / 192.f);
  float dv = v - mu;
  float var = block_reduce_192(dv * dv, red, c) * (1.f / 192.f);
  out[row * C_DIM + c] = (h16)(dv * rsqrtf(var + 1e-5f) * g[c] + b[c]);
}

// ================= QKV GEMM: scatter to per-head Q (scaled) / K / V ================
__global__ __launch_bounds__(128) void gemm_qkv(const h16* __restrict__ A,
    const h16* __restrict__ Bw, const float* __restrict__ bias,
    h16* __restrict__ q, h16* __restrict__ k, h16* __restrict__ v) {
  v8f acc[4];
  int lane = threadIdx.x & 31, wave = threadIdx.x >> 5;
  size_t m0 = (size_t)blockIdx.x * 64 + wave * 16;
  int n0 = blockIdx.y * 64;
  gemm_rows16(A, C_DIM, Bw, 576, C_DIM, m0, n0, lane, acc);
  int mb = 8 * (lane >> 4), nc = lane & 15;
#pragma unroll
  for (int nt = 0; nt < 4; ++nt) {
    int col = n0 + nt * 16 + nc;
    int which = col / 192, head = (col % 192) / 32, dd = col & 31;
    float bia = bias[col];
    h16* dst = (which == 0) ? q : ((which == 1) ? k : v);
#pragma unroll
    for (int r = 0; r < 8; ++r) {
      size_t gm = m0 + mb + r;
      int win = (int)(gm / NPAD), n = (int)(gm % NPAD);
      float val = acc[nt][r] + bia;
      if (which == 0) val *= 0.17677669529663689f;   // 1/sqrt(32)
      dst[(((size_t)win * N_HEADS + head) * NPAD + n) * 32 + dd] = (h16)val;
    }
  }
}

// ================= fused attention: one workgroup per (window, head) ===============
__global__ __launch_bounds__(128) void attn_kernel(const h16* __restrict__ Qm,
    const h16* __restrict__ Km, const h16* __restrict__ Vm,
    const float* __restrict__ table, h16* __restrict__ AO, int shifted) {
  __shared__ __align__(16) h16 Qs[NPAD * 32], Ks[NPAD * 32];
  __shared__ __align__(16) h16 Vsw[128 * 32];       // swizzled [kb][n][kh][e]
  __shared__ __align__(16) h16 P[NPAD * 128];
  __shared__ float tabH[507];
  __shared__ int cnt[NPAD];
  int wg = blockIdx.x, win = wg / N_HEADS, head = wg % N_HEADS;
  int tid = threadIdx.x, lane = tid & 31, wave = tid >> 5;
  for (int i = tid; i < 507; i += 128) tabH[i] = table[i * 6 + head];
  int wd = (win >> 6) & 7, wh = (win >> 3) & 7, ww = win & 7;
  for (int i = tid; i < NPAD; i += 128) {
    int cv = 0;
    if (shifted && i < NTOK) {
      int td = i / 49, rem = i % 49, th = rem / 7, tw = rem % 7;
      int d = wd * 2 + td, h = wh * 7 + th, w = ww * 7 + tw;
      int rd = (d < 14) ? 0 : ((d < 15) ? 1 : 2);
      int rh = (h < 49) ? 0 : ((h < 53) ? 1 : 2);
      int rw = (w < 49) ? 0 : ((w < 53) ? 1 : 2);
      cv = rd * 9 + rh * 3 + rw;
    }
    cnt[i] = cv;
  }
  const h16* Qg = Qm + ((size_t)win * N_HEADS + head) * NPAD * 32;
  const h16* Kg = Km + ((size_t)win * N_HEADS + head) * NPAD * 32;
  const h16* Vg = Vm + ((size_t)win * N_HEADS + head) * NPAD * 32;
  for (int i = tid; i < 448; i += 128) {            // 112x32 halfs as 16B chunks
    ((u32x4*)Qs)[i] = ((const u32x4*)Qg)[i];
    ((u32x4*)Ks)[i] = ((const u32x4*)Kg)[i];
  }
  for (int i = tid; i < 4096; i += 128) {           // V swizzled; zero K-pad rows
    int kk2 = i >> 5, n = i & 31;
    h16 val = (kk2 < NPAD) ? Vg[kk2 * 32 + n] : (h16)0.f;
    Vsw[(((kk2 >> 5) * 32 + n) * 2 + ((kk2 >> 4) & 1)) * 16 + (kk2 & 15)] = val;
  }
  __syncthreads();

  int nc = lane & 15, hb = lane >> 4;
  // S = Q*K^T + bias + mask, softmax -> P (f16, 112x128)
  for (int ti = wave; ti < 7; ti += 4) {
    v8f s[7];
    v16h af = a_frag(Qs + ti * 16 * 32, 32, lane);
#pragma unroll
    for (int tj = 0; tj < 7; ++tj) {
      // B[k][n] = K[tj*16+n][k]: contiguous 32B run per (lane, half)
      v16h bf = c_frag32(Ks + (tj * 16 + nc) * 32 + hb * 16);
      v8f z;
#pragma unroll
      for (int r = 0; r < 8; ++r) z[r] = 0.f;
      s[tj] = __builtin_amdgcn_wmma_f32_16x16x32_f16(false, af, false, bf, (short)0, z,
                                                     false, false);
    }
#pragma unroll
    for (int tj = 0; tj < 7; ++tj) {
      int j = tj * 16 + nc;
#pragma unroll
      for (int r = 0; r < 8; ++r) {
        int m = ti * 16 + r + 8 * hb;
        float val = s[tj][r];
        if (m < NTOK) {
          if (j < NTOK) {
            int ad = m / 49, ah = (m % 49) / 7, aw = m % 7;
            int bd = j / 49, bh = (j % 49) / 7, bw = j % 7;
            int idx = ((ad - bd + 1) * 13 + (ah - bh + 6)) * 13 + (aw - bw + 6);
            val += tabH[idx];
            if (shifted && cnt[m] != cnt[j]) val += 2.f;   // reference mask value
          } else {
            val = -1e9f;                                    // padded key columns
          }
        }
        s[tj][r] = val;
      }
    }
#pragma unroll
    for (int r = 0; r < 8; ++r) {
      float mx = -1e30f;
#pragma unroll
      for (int tj = 0; tj < 7; ++tj) mx = fmaxf(mx, s[tj][r]);
      for (int off = 1; off < 16; off <<= 1) mx = fmaxf(mx, __shfl_xor(mx, off));
      float e[7], sm = 0.f;
#pragma unroll
      for (int tj = 0; tj < 7; ++tj) { e[tj] = __expf(s[tj][r] - mx); sm += e[tj]; }
      for (int off = 1; off < 16; off <<= 1) sm += __shfl_xor(sm, off);
      float inv = 1.f / sm;
      int m = ti * 16 + r + 8 * hb;
      float keep = (m < NTOK) ? inv : 0.f;
#pragma unroll
      for (int tj = 0; tj < 7; ++tj) P[m * 128 + tj * 16 + nc] = (h16)(e[tj] * keep);
      P[m * 128 + 112 + nc] = (h16)0.f;                     // zero K-pad columns
    }
  }
  __syncthreads();
  // O = P(112x128) * V(128x32)
  for (int t = wave; t < 14; t += 4) {
    int ti = t >> 1, tj = t & 1;
    v8f acc;
#pragma unroll
    for (int r = 0; r < 8; ++r) acc[r] = 0.f;
#pragma unroll
    for (int kk = 0; kk < 4; ++kk) {
      v16h af = a_frag(P + ti * 16 * 128 + kk * 32, 128, lane);
      v16h bf = c_frag32(Vsw + ((size_t)((kk * 32 + tj * 16 + nc) * 2 + hb)) * 16);
      acc = __builtin_amdgcn_wmma_f32_16x16x32_f16(false, af, false, bf, (short)0, acc,
                                                   false, false);
    }
#pragma unroll
    for (int r = 0; r < 8; ++r) {
      int m = ti * 16 + r + 8 * hb;
      AO[((size_t)win * NPAD + m) * C_DIM + head * 32 + tj * 16 + nc] = (h16)acc[r];
    }
  }
}

// ================= proj GEMM + window reverse + un-shift + residual ================
__global__ __launch_bounds__(128) void gemm_proj(const h16* __restrict__ A,
    const h16* __restrict__ Bw, const float* __restrict__ bias,
    const float* __restrict__ xin, float* __restrict__ out, int shifted) {
  v8f acc[4];
  int lane = threadIdx.x & 31, wave = threadIdx.x >> 5;
  size_t m0 = (size_t)blockIdx.x * 64 + wave * 16;
  int n0 = blockIdx.y * 64;
  gemm_rows16(A, C_DIM, Bw, C_DIM, C_DIM, m0, n0, lane, acc);
  int mb = 8 * (lane >> 4), nc = lane & 15;
#pragma unroll
  for (int nt = 0; nt < 4; ++nt) {
    int col = n0 + nt * 16 + nc;
    float bia = bias[col];
#pragma unroll
    for (int r = 0; r < 8; ++r) {
      size_t gm = m0 + mb + r;
      int win = (int)(gm / NPAD), n = (int)(gm % NPAD);
      if (n < NTOK) {
        int bb = win >> 9, wd = (win >> 6) & 7, wh = (win >> 3) & 7, ww = win & 7;
        int td = n / 49, rem = n % 49, th = rem / 7, tw = rem % 7;
        int d = wd * 2 + td, h = wh * 7 + th, w = ww * 7 + tw;
        if (shifted) { d = (d + 1) & 15; h = (h + 3) % 56; w = (w + 3) % 56; }
        size_t tok = (((size_t)bb * 16 + d) * 56 + h) * 56 + w;
        out[tok * C_DIM + col] = xin[tok * C_DIM + col] + acc[nt][r] + bia;
      }
    }
  }
}

// ================= FC1 GEMM + exact GELU -> f16 ====================================
__global__ __launch_bounds__(128) void gemm_fc1(const h16* __restrict__ A,
    const h16* __restrict__ Bw, const float* __restrict__ bias, h16* __restrict__ out) {
  v8f acc[4];
  int lane = threadIdx.x & 31, wave = threadIdx.x >> 5;
  size_t m0 = (size_t)blockIdx.x * 64 + wave * 16;
  int n0 = blockIdx.y * 64;
  gemm_rows16(A, C_DIM, Bw, HID, C_DIM, m0, n0, lane, acc);
  int mb = 8 * (lane >> 4), nc = lane & 15;
#pragma unroll
  for (int nt = 0; nt < 4; ++nt) {
    int col = n0 + nt * 16 + nc;
    float bia = bias[col];
#pragma unroll
    for (int r = 0; r < 8; ++r) {
      float v = acc[nt][r] + bia;
      v = 0.5f * v * (1.f + erff(v * 0.70710678118654752f));
      out[(m0 + mb + r) * (size_t)HID + col] = (h16)v;
    }
  }
}

// ================= FC2 GEMM + residual add into d_out ==============================
__global__ __launch_bounds__(128) void gemm_fc2(const h16* __restrict__ A,
    const h16* __restrict__ Bw, const float* __restrict__ bias, float* __restrict__ out) {
  v8f acc[4];
  int lane = threadIdx.x & 31, wave = threadIdx.x >> 5;
  size_t m0 = (size_t)blockIdx.x * 64 + wave * 16;
  int n0 = blockIdx.y * 64;
  gemm_rows16(A, HID, Bw, C_DIM, HID, m0, n0, lane, acc);
  int mb = 8 * (lane >> 4), nc = lane & 15;
#pragma unroll
  for (int nt = 0; nt < 4; ++nt) {
    int col = n0 + nt * 16 + nc;
    float bia = bias[col];
#pragma unroll
    for (int r = 0; r < 8; ++r) {
      size_t idx = (m0 + mb + r) * (size_t)C_DIM + col;
      out[idx] = out[idx] + acc[nt][r] + bia;
    }
  }
}

// ===================================================================================
extern "C" void kernel_launch(void* const* d_in, const int* in_sizes, int n_in,
                              void* d_out, int out_size, void* d_ws, size_t ws_size,
                              hipStream_t stream) {
  (void)in_sizes; (void)n_in; (void)out_size; (void)ws_size;
  const float* X    = (const float*)d_in[0];
  const float* LN1G = (const float*)d_in[1];
  const float* LN1B = (const float*)d_in[2];
  const float* QKVW = (const float*)d_in[3];
  const float* QKVB = (const float*)d_in[4];
  const float* PROJW= (const float*)d_in[5];
  const float* PROJB= (const float*)d_in[6];
  const float* TBL  = (const float*)d_in[7];
  const float* LN2G = (const float*)d_in[8];
  const float* LN2B = (const float*)d_in[9];
  const float* FC1W = (const float*)d_in[10];
  const float* FC1B = (const float*)d_in[11];
  const float* FC2W = (const float*)d_in[12];
  const float* FC2B = (const float*)d_in[13];
  float* OUT = (float*)d_out;
  char* ws = (char*)d_ws;
  h16* X16  = (h16*)(ws + OFF_X16);
  h16* Qb   = (h16*)(ws + OFF_Q);
  h16* Kb   = (h16*)(ws + OFF_K);
  h16* Vb   = (h16*)(ws + OFF_V);
  h16* AOb  = (h16*)(ws + OFF_AO);
  h16* LN2b = (h16*)(ws + OFF_LN2);
  h16* FC1b = (h16*)(ws + OFF_FC1);
  h16* Wqkv = (h16*)(ws + OFF_WQKV);
  h16* Wproj= (h16*)(ws + OFF_WPROJ);
  h16* Wfc1 = (h16*)(ws + OFF_WFC1);
  h16* Wfc2 = (h16*)(ws + OFF_WFC2);

  for (int blk = 0; blk < 2; ++blk) {
    int shifted = blk;                           // block 1 is shifted
    const float* xin = (blk == 0) ? X : (const float*)OUT;
    // weight convert/transpose to fragment-swizzled f16
    convert_wT_sw<<<(192 * 576 + 255) / 256, 256, 0, stream>>>(QKVW + (size_t)blk * 576 * 192, Wqkv, 192, 576);
    convert_wT_sw<<<(192 * 192 + 255) / 256, 256, 0, stream>>>(PROJW + (size_t)blk * 192 * 192, Wproj, 192, 192);
    convert_wT_sw<<<(192 * 768 + 255) / 256, 256, 0, stream>>>(FC1W + (size_t)blk * 768 * 192, Wfc1, 192, 768);
    convert_wT_sw<<<(768 * 192 + 255) / 256, 256, 0, stream>>>(FC2W + (size_t)blk * 192 * 768, Wfc2, 768, 192);
    // LN1 + shift + window partition
    ln_part_kernel<<<MROWS, 192, 0, stream>>>(xin, LN1G + blk * 192, LN1B + blk * 192, X16, shifted);
    // QKV projection
    gemm_qkv<<<dim3(MROWS / 64, 576 / 64), 128, 0, stream>>>(X16, Wqkv, QKVB + blk * 576, Qb, Kb, Vb);
    // fused windowed attention
    attn_kernel<<<NWIN * N_HEADS, 128, 0, stream>>>(Qb, Kb, Vb, TBL + (size_t)blk * 507 * 6, AOb, shifted);
    // output proj + window reverse + un-shift + residual (writes running x into d_out)
    gemm_proj<<<dim3(MROWS / 64, 192 / 64), 128, 0, stream>>>(AOb, Wproj, PROJB + blk * 192, xin, OUT, shifted);
    // MLP
    ln2_kernel<<<TOKS, 192, 0, stream>>>((const float*)OUT, LN2G + blk * 192, LN2B + blk * 192, LN2b);
    gemm_fc1<<<dim3(TOKS / 64, HID / 64), 128, 0, stream>>>(LN2b, Wfc1, FC1B + blk * HID, FC1b);
    gemm_fc2<<<dim3(TOKS / 64, 192 / 64), 128, 0, stream>>>(FC1b, Wfc2, FC2B + blk * 192, OUT);
  }
}
